// MultiHeadAttention_18056042513018
// MI455X (gfx1250) — matrix operand order repaired and checked by reading, not compiled, not run on hardware
//
#include <hip/hip_runtime.h>
#include <hip/hip_bf16.h>

// ---------------------------------------------------------------------------
// MHA forward on gfx1250 (MI455X): bf16 WMMA, f32 accumulate, async-LDS staging.
//   B=2, S=2048, D=1024, H=16, DEPTH=64
// Workspace layout (bytes), total 64 MB:
//   qb @0, kb @8MB, vb @16MB, ctx @24MB, wqt/wkt/wvt/wot @32MB (2MB each, N-major),
//   Qh @40MB [B,H,S,64], Kh @48MB, Vt @56MB [B,H,64,S]
// ---------------------------------------------------------------------------

#define BB 2
#define SS 2048
#define DD 1024
#define HH 16
#define DEPTH 64
#define MM (BB * SS)   // 4096

typedef __attribute__((ext_vector_type(16))) __bf16 v16bf;
typedef __attribute__((ext_vector_type(8)))  float  v8f;
typedef __attribute__((ext_vector_type(4)))  unsigned int v4u;
typedef __attribute__((ext_vector_type(4)))  float  v4f;
typedef __attribute__((ext_vector_type(2)))  unsigned int v2u;

// ---- fragment loaders (16-bit, wave32; contiguous k order, same in A and B) ------
// A 16x32: lane m=l16, half g: elems i -> k = k0 + g*16 + i (contiguous)
__device__ __forceinline__ v16bf load_a_frag(const __bf16* __restrict__ base,
                                             int ld, int m0, int k0,
                                             int l16, int g) {
  const __bf16* p = base + (size_t)(m0 + l16) * ld + k0 + g * 16;
  union { v4u u[2]; v16bf v; } t;
  t.u[0] = *(const v4u*)p;
  t.u[1] = *(const v4u*)(p + 8);
  return t.v;
}
// B 32x16: lane n=n0+l16, half g: elems i -> k = k0 + g*16 + i (contiguous)
__device__ __forceinline__ v16bf load_b_frag(const __bf16* __restrict__ base,
                                             int ld, int n0, int k0,
                                             int l16, int g) {
  const __bf16* p = base + (size_t)(n0 + l16) * ld + k0 + g * 16;
  union { v4u u[2]; v16bf v; } t;
  t.u[0] = *(const v4u*)p;
  t.u[1] = *(const v4u*)(p + 8);
  return t.v;
}

// ---- converters (4-wide) --------------------------------------------------
__global__ void cvt_bf16_kernel(const float* __restrict__ src,
                                __bf16* __restrict__ dst, int n4) {
  int i = blockIdx.x * blockDim.x + threadIdx.x;
  if (i >= n4) return;
  v4f f = *(const v4f*)(src + (size_t)i * 4);
  union { __bf16 b[4]; v2u u; } t;
  t.b[0] = (__bf16)f.x; t.b[1] = (__bf16)f.y;
  t.b[2] = (__bf16)f.z; t.b[3] = (__bf16)f.w;
  *(v2u*)(dst + (size_t)i * 4) = t.u;
}

// w[k*D+n] f32 -> wt[n*D+k] bf16 (transpose so B-fragments are contiguous)
__global__ void cvt_tr_bf16_kernel(const float* __restrict__ w,
                                   __bf16* __restrict__ wt) {
  int i = blockIdx.x * blockDim.x + threadIdx.x;   // 0 .. D*D/4-1
  int k  = i >> 8;                                  // i*4 / 1024
  int n0 = (i & 255) << 2;
  v4f f = *(const v4f*)(w + (size_t)k * DD + n0);
  wt[(size_t)(n0 + 0) * DD + k] = (__bf16)f.x;
  wt[(size_t)(n0 + 1) * DD + k] = (__bf16)f.y;
  wt[(size_t)(n0 + 2) * DD + k] = (__bf16)f.z;
  wt[(size_t)(n0 + 3) * DD + k] = (__bf16)f.w;
}

// ---- projection GEMM: Y[4096,1024] = X(bf16) @ Wt(bf16,N-major) + bias ----
// B tile (64n x 32k) staged in LDS via async copy, ping-pong buffered with a
// manual 2x unroll (no loop-carried register rotation); A frags register
// double-buffered. 4 waves/block, each a 16x64 strip.
// mode 0: bf16 out [B,H,S,64] | mode 1: bf16 out [B,H,64,S] | mode 2: f32 [M,D]
__global__ __launch_bounds__(128)
void gemm_proj_kernel(const __bf16* __restrict__ X,
                      const __bf16* __restrict__ Wt,
                      const float* __restrict__ bias,
                      void* __restrict__ out, int mode) {
  __shared__ __align__(16) __bf16 bt[2][64][32];   // 8 KB ping-pong B tile

  const int tid  = threadIdx.x;
  const int lane = tid & 31;
  const int wv   = tid >> 5;
  const int l16  = lane & 15;
  const int g    = lane >> 4;
  const int m0   = blockIdx.y * 64 + wv * 16;
  const int n0   = blockIdx.x * 64;

  // async copy: thread tid moves 32B of the 4KB tile (2 x b128; the 24-bit
  // IOFFSET is added to BOTH the LDS and the global address, ISA §15.18).
  const int crow = tid >> 1;              // 0..63  (n within tile)
  const int ccol = (tid & 1) << 5;        // 0 or 32 bytes (k*2)
  const __bf16* grow = Wt + (size_t)(n0 + crow) * DD + (ccol >> 1);

  // lds_off is ptrtoint of &bt[...]: low 32 bits of an LDS object's flat
  // address are the wave-relative LDS byte offset (ISA §10.2); the pointer
  // capture + "memory" clobber keeps the ds_load fragment reads live.
  auto issue_copy = [&](int k0, int buf) {
    unsigned lds_off = (unsigned)(size_t)&bt[buf][crow][ccol >> 1];
    unsigned long long ga = (unsigned long long)(grow + k0);
    asm volatile(
        "global_load_async_to_lds_b128 %0, %1, off\n\t"
        "global_load_async_to_lds_b128 %0, %1, off offset:16"
        :: "v"(lds_off), "v"(ga) : "memory");
  };

  // one k-step: batch all 4 B-fragment LDS reads, then 4 WMMAs
  v8f acc[4] = {};
  auto do_step = [&](int buf, v16bf a) {
    v16bf bfr[4];
#pragma unroll
    for (int c = 0; c < 4; ++c) {
      const __bf16* p = &bt[buf][c * 16 + l16][g * 16];
      union { v4u u[2]; v16bf v; } t;
      t.u[0] = *(const v4u*)p;
      t.u[1] = *(const v4u*)(p + 8);
      bfr[c] = t.v;
    }
#pragma unroll
    for (int c = 0; c < 4; ++c)
      acc[c] = __builtin_amdgcn_wmma_f32_16x16x32_bf16(
          false, a, false, bfr[c], (short)0, acc[c], false, false);
  };

  constexpr int NSTEP = DD / 32;          // 32 (even)
  issue_copy(0, 0);
  v16bf a0 = load_a_frag(X, DD, m0, 0, l16, g);
  v16bf a1 = a0;
  asm volatile("s_wait_asynccnt 0x0" ::: "memory");
  __syncthreads();

  for (int step = 0; step < NSTEP; step += 2) {
    // even step: consume bt[0], prefetch step+1 into bt[1]
    if (step + 1 < NSTEP) {
      issue_copy((step + 1) * 32, 1);
      a1 = load_a_frag(X, DD, m0, (step + 1) * 32, l16, g);
    }
    do_step(0, a0);
    asm volatile("s_wait_asynccnt 0x0" ::: "memory");
    __syncthreads();

    // odd step: consume bt[1], prefetch step+2 into bt[0]
    if (step + 2 < NSTEP) {
      issue_copy((step + 2) * 32, 0);
      a0 = load_a_frag(X, DD, m0, (step + 2) * 32, l16, g);
    }
    do_step(1, a1);
    asm volatile("s_wait_asynccnt 0x0" ::: "memory");
    __syncthreads();
  }

#pragma unroll
  for (int c = 0; c < 4; ++c) {
    const int n = n0 + c * 16 + l16;
    const float bv = bias[n];
#pragma unroll
    for (int r = 0; r < 8; ++r) {
      const int m = m0 + r + 8 * g;
      const float val = acc[c][r] + bv;
      if (mode == 2) {
        ((float*)out)[(size_t)m * DD + n] = val;
      } else {
        const int bi = m >> 11;
        const int si = m & (SS - 1);
        const int hi = n >> 6;
        const int di = n & (DEPTH - 1);
        __bf16* o = (__bf16*)out;
        if (mode == 0)
          o[(((size_t)bi * HH + hi) * SS + si) * DEPTH + di] = (__bf16)val;
        else
          o[(((size_t)bi * HH + hi) * DEPTH + di) * SS + si] = (__bf16)val;
      }
    }
  }
}

// ---- flash attention: per (b,h), 64 queries/block (16/wave), keys x32 -----
__global__ __launch_bounds__(128)
void attn_kernel(const __bf16* __restrict__ Qh,
                 const __bf16* __restrict__ Kh,
                 const __bf16* __restrict__ Vt,
                 __bf16* __restrict__ ctx) {
  const int bh   = blockIdx.y;
  const int b    = bh / HH;
  const int h    = bh % HH;
  const int lane = threadIdx.x & 31;
  const int wv   = threadIdx.x >> 5;
  const int l16  = lane & 15;
  const int g    = lane >> 4;
  const int q0   = blockIdx.x * 64 + wv * 16;

  const __bf16* Qb = Qh + (size_t)bh * SS * DEPTH;
  const __bf16* Kb = Kh + (size_t)bh * SS * DEPTH;
  const __bf16* Vb = Vt + (size_t)bh * DEPTH * SS;

  __shared__ __align__(16) __bf16 lds_p[4][16][32];  // per-wave P tile

  v16bf qf0 = load_a_frag(Qb, DEPTH, q0, 0,  l16, g);
  v16bf qf1 = load_a_frag(Qb, DEPTH, q0, 32, l16, g);

  // K fragments for tile 0 (register double-buffered across the key loop)
  v16bf kf[4];
  kf[0] = load_b_frag(Kb, DEPTH, 0,  0,  l16, g);
  kf[1] = load_b_frag(Kb, DEPTH, 16, 0,  l16, g);
  kf[2] = load_b_frag(Kb, DEPTH, 0,  32, l16, g);
  kf[3] = load_b_frag(Kb, DEPTH, 16, 32, l16, g);

  v8f acc[4] = {};
  float mrun[8], lrun[8];
#pragma unroll
  for (int r = 0; r < 8; ++r) { mrun[r] = -1e30f; lrun[r] = 0.f; }

  const float LOG2E = 1.44269504088896f;
  const float SCALE = 0.125f;            // 1/sqrt(64)

  for (int kt = 0; kt < SS; kt += 32) {
    // issue V-fragment loads early: latency hides under scores + softmax
    v16bf vf[4];
#pragma unroll
    for (int c = 0; c < 4; ++c) vf[c] = load_b_frag(Vb, SS, c * 16, kt, l16, g);

    // scores: 16 queries x 32 keys, K=depth=64
    v8f s0 = {}, s1 = {};
    s0 = __builtin_amdgcn_wmma_f32_16x16x32_bf16(false, qf0, false, kf[0], (short)0, s0, false, false);
    s1 = __builtin_amdgcn_wmma_f32_16x16x32_bf16(false, qf0, false, kf[1], (short)0, s1, false, false);
    s0 = __builtin_amdgcn_wmma_f32_16x16x32_bf16(false, qf1, false, kf[2], (short)0, s0, false, false);
    s1 = __builtin_amdgcn_wmma_f32_16x16x32_bf16(false, qf1, false, kf[3], (short)0, s1, false, false);

    // prefetch next key tile's K fragments (overlaps softmax VALU)
    if (kt + 32 < SS) {
      kf[0] = load_b_frag(Kb, DEPTH, kt + 32, 0,  l16, g);
      kf[1] = load_b_frag(Kb, DEPTH, kt + 48, 0,  l16, g);
      kf[2] = load_b_frag(Kb, DEPTH, kt + 32, 32, l16, g);
      kf[3] = load_b_frag(Kb, DEPTH, kt + 48, 32, l16, g);
    }

    // online softmax (row stats across 16-lane halves; denominator kept partial)
#pragma unroll
    for (int r = 0; r < 8; ++r) {
      float a0 = s0[r] * SCALE;
      float a1 = s1[r] * SCALE;
      float t  = fmaxf(a0, a1);
      t = fmaxf(t, __shfl_xor(t, 1));
      t = fmaxf(t, __shfl_xor(t, 2));
      t = fmaxf(t, __shfl_xor(t, 4));
      t = fmaxf(t, __shfl_xor(t, 8));
      float mnew = fmaxf(mrun[r], t);
      float corr = exp2f((mrun[r] - mnew) * LOG2E);
      mrun[r] = mnew;
      lrun[r] *= corr;
      acc[0][r] *= corr; acc[1][r] *= corr;
      acc[2][r] *= corr; acc[3][r] *= corr;
      float p0 = exp2f((a0 - mnew) * LOG2E);
      float p1 = exp2f((a1 - mnew) * LOG2E);
      lrun[r] += p0 + p1;
      lds_p[wv][r + 8 * g][l16]      = (__bf16)p0;  // C-layout -> row-major LDS
      lds_p[wv][r + 8 * g][l16 + 16] = (__bf16)p1;
    }
    // intra-wave LDS transpose: lockstep wave, drain DS then re-read
    asm volatile("s_wait_dscnt 0x0" ::: "memory");

    v16bf pa;
    {
      const __bf16* pp = &lds_p[wv][l16][g * 16];
      union { v4u u[2]; v16bf v; } t;
      t.u[0] = *(const v4u*)pp;
      t.u[1] = *(const v4u*)(pp + 8);
      pa = t.v;
    }

    // out += P(16x32) @ V(32x64)
#pragma unroll
    for (int c = 0; c < 4; ++c)
      acc[c] = __builtin_amdgcn_wmma_f32_16x16x32_bf16(
          false, pa, false, vf[c], (short)0, acc[c], false, false);

    asm volatile("s_wait_dscnt 0x0" ::: "memory");  // reads done before overwrite
  }

  // finalize: reduce denominator, normalize, write ctx [B,S,D] bf16
#pragma unroll
  for (int r = 0; r < 8; ++r) {
    float l = lrun[r];
    l += __shfl_xor(l, 1);
    l += __shfl_xor(l, 2);
    l += __shfl_xor(l, 4);
    l += __shfl_xor(l, 8);
    const float inv = 1.0f / l;
    const int row = q0 + r + 8 * g;
    const size_t base = ((size_t)b * SS + row) * DD + h * DEPTH;
#pragma unroll
    for (int c = 0; c < 4; ++c)
      ctx[base + c * 16 + l16] = (__bf16)(acc[c][r] * inv);
  }
}

// ---------------------------------------------------------------------------
extern "C" void kernel_launch(void* const* d_in, const int* in_sizes, int n_in,
                              void* d_out, int out_size, void* d_ws, size_t ws_size,
                              hipStream_t stream) {
  const float* q  = (const float*)d_in[0];
  const float* k  = (const float*)d_in[1];
  const float* v  = (const float*)d_in[2];
  const float* wq = (const float*)d_in[3];
  const float* bq = (const float*)d_in[4];
  const float* wk = (const float*)d_in[5];
  const float* bk = (const float*)d_in[6];
  const float* wv = (const float*)d_in[7];
  const float* bv = (const float*)d_in[8];
  const float* wo = (const float*)d_in[9];
  const float* bo = (const float*)d_in[10];

  char* ws = (char*)d_ws;
  const size_t SZ_BSD = (size_t)BB * SS * DD * sizeof(__bf16);  // 8 MB
  const size_t SZ_DDb = (size_t)DD * DD * sizeof(__bf16);       // 2 MB
  __bf16* qb  = (__bf16*)(ws);
  __bf16* kb  = (__bf16*)(ws + SZ_BSD);
  __bf16* vb  = (__bf16*)(ws + 2 * SZ_BSD);
  __bf16* ctx = (__bf16*)(ws + 3 * SZ_BSD);
  __bf16* wqt = (__bf16*)(ws + 4 * SZ_BSD);
  __bf16* wkt = (__bf16*)(ws + 4 * SZ_BSD + SZ_DDb);
  __bf16* wvt = (__bf16*)(ws + 4 * SZ_BSD + 2 * SZ_DDb);
  __bf16* wot = (__bf16*)(ws + 4 * SZ_BSD + 3 * SZ_DDb);
  __bf16* Qh  = (__bf16*)(ws + 4 * SZ_BSD + 4 * SZ_DDb);
  __bf16* Kh  = (__bf16*)(ws + 5 * SZ_BSD + 4 * SZ_DDb);
  __bf16* Vt  = (__bf16*)(ws + 6 * SZ_BSD + 4 * SZ_DDb);

  const int nBSD4 = BB * SS * DD / 4;  // 1,048,576 vec4s
  const int nDD4  = DD * DD / 4;       // 262,144 vec4s

  cvt_bf16_kernel<<<nBSD4 / 256, 256, 0, stream>>>(q, qb, nBSD4);
  cvt_bf16_kernel<<<nBSD4 / 256, 256, 0, stream>>>(k, kb, nBSD4);
  cvt_bf16_kernel<<<nBSD4 / 256, 256, 0, stream>>>(v, vb, nBSD4);
  cvt_tr_bf16_kernel<<<nDD4 / 256, 256, 0, stream>>>(wq, wqt);
  cvt_tr_bf16_kernel<<<nDD4 / 256, 256, 0, stream>>>(wk, wkt);
  cvt_tr_bf16_kernel<<<nDD4 / 256, 256, 0, stream>>>(wv, wvt);
  cvt_tr_bf16_kernel<<<nDD4 / 256, 256, 0, stream>>>(wo, wot);

  dim3 gemm_grid(DD / 64, MM / 64);   // (16, 64)
  gemm_proj_kernel<<<gemm_grid, 128, 0, stream>>>(qb, wqt, bq, (void*)Qh, 0);
  gemm_proj_kernel<<<gemm_grid, 128, 0, stream>>>(kb, wkt, bk, (void*)Kh, 0);
  gemm_proj_kernel<<<gemm_grid, 128, 0, stream>>>(vb, wvt, bv, (void*)Vt, 1);

  dim3 attn_grid(SS / 64, BB * HH);   // (32, 32)
  attn_kernel<<<attn_grid, 128, 0, stream>>>(Qh, Kh, Vt, ctx);

  gemm_proj_kernel<<<gemm_grid, 128, 0, stream>>>(ctx, wot, bo, d_out, 2);
}
